// Seq2Seq_4492535792260
// MI455X (gfx1250) — matrix operand order repaired and checked by reading, not compiled, not run on hardware
//
#include <hip/hip_runtime.h>
#include <hip/hip_bf16.h>

// ---------------------------------------------------------------------------
// Seq2Seq 2-layer LSTM (enc T=512 + dec T=512) + final linear, MI455X gfx1250.
// Persistent-kernel design: 64 blocks x 256 threads = 512 wave32 waves.
//   wave id -> (layer, mtile(0..3), htile(0..63)); each wave owns a 16x16
//   (batch x hidden) patch and its 4 gate tiles (i,f,g,o).
// Cell state c stays in v8f registers (WMMA C/D layout) for all 1024 steps.
// h is double-buffered (step parity) as bf16 row-major [64][1024] -- exactly
// the A-matrix operand for the next step's WMMAs.
// Layer wavefront: at super-step s, layer0 computes t=s while layer1 computes
// t=s-1; one grid barrier per super-step.
// ---------------------------------------------------------------------------

typedef __bf16 bf16;
typedef __attribute__((ext_vector_type(16))) __bf16 bf16x16;
typedef __attribute__((ext_vector_type(8)))  __bf16 bf16x8;
typedef __attribute__((ext_vector_type(8)))  float  f32x8;

#define B_ 64
#define T_ 512
#define I_ 128
#define H_ 1024
#define O_ 128
#define NBLK 64
#define NTHREADS 256

__device__ __forceinline__ f32x8 wmma_bf16(bf16x16 a, bf16x16 b, f32x8 c) {
  return __builtin_amdgcn_wmma_f32_16x16x32_bf16(false, a, false, b,
                                                 (short)0, c, false, false);
}

// A fragment, 16x32 bf16 from row-major [M, lda] memory.
// ISA 7.12.2: lanes 0-15 hold row M=lane, K={0..7,16..23}; lanes 16-31 hold
// row M=lane-16, K={8..15,24..31}. Two 16B chunks per lane.
__device__ __forceinline__ bf16x16 load_a_frag(const bf16* __restrict__ A,
                                               size_t lda, int row0, int k,
                                               int lane) {
  const int r = lane & 15, hi = lane >> 4;
  const bf16* p = A + (size_t)(row0 + r) * lda + k + hi * 8;
  bf16x8 lo = *(const bf16x8*)p;        // K base+{0..7} or {8..15}
  bf16x8 hc = *(const bf16x8*)(p + 16); // K base+{16..23} or {24..31}
  bf16x16 out;
#pragma unroll
  for (int i = 0; i < 8; ++i) { out[i] = lo[i]; out[i + 8] = hc[i]; }
  return out;
}

// B fragment, 32x16 bf16 where memory holds W[N, ldw] row-major and
// B[k,n] = W[col0+n, k]. Lanes 0-15 = column n with K={0..7,16..23}, lanes
// 16-31 = same columns with K={8..15,24..31} (matches A) -> two 16B loads
// per lane from row (col0 + n).
__device__ __forceinline__ bf16x16 load_b_frag(const bf16* __restrict__ W,
                                               size_t ldw, int col0, int k,
                                               int lane) {
  const int nn = lane & 15, hi = lane >> 4;
  const bf16* p = W + (size_t)(col0 + nn) * ldw + k + hi * 8;
  bf16x8 lo = *(const bf16x8*)p;
  bf16x8 hc = *(const bf16x8*)(p + 16);
  bf16x16 out;
#pragma unroll
  for (int i = 0; i < 8; ++i) { out[i] = lo[i]; out[i + 8] = hc[i]; }
  return out;
}

// Accumulate 4 gate tiles: acc[g] += A[mtile] x W[g*H + htile*16 cols]^T
__device__ __forceinline__ void gemm4(f32x8 acc[4], const bf16* __restrict__ A,
                                      size_t lda, const bf16* __restrict__ W,
                                      size_t ldw, int K, int mtile, int htile,
                                      int lane) {
  for (int k = 0; k < K; k += 32) {
    bf16x16 a = load_a_frag(A, lda, mtile * 16, k, lane);
#pragma unroll
    for (int g = 0; g < 4; ++g) {
      bf16x16 b = load_b_frag(W, ldw, g * H_ + htile * 16, k, lane);
      acc[g] = wmma_bf16(a, b, acc[g]);
    }
  }
}

__device__ __forceinline__ void grid_barrier(unsigned* cnt, unsigned& gen) {
  __threadfence();
  __syncthreads();
  if (threadIdx.x == 0) {
    __hip_atomic_fetch_add(cnt, 1u, __ATOMIC_ACQ_REL, __HIP_MEMORY_SCOPE_AGENT);
    const unsigned tgt = (gen + 1) * NBLK;
    while (__hip_atomic_load(cnt, __ATOMIC_ACQUIRE, __HIP_MEMORY_SCOPE_AGENT) < tgt)
      __builtin_amdgcn_s_sleep(2);
  }
  ++gen;
  __syncthreads();
  __threadfence();
}

__device__ __forceinline__ float sigm(float x) {
  return 1.0f / (1.0f + __expf(-x));
}

__global__ __launch_bounds__(NTHREADS, 1) void lstm_seq2seq(
    const bf16* __restrict__ Xbf,
    const bf16* __restrict__ eWih0, const bf16* __restrict__ eWhh0,
    const bf16* __restrict__ eWih1, const bf16* __restrict__ eWhh1,
    const bf16* __restrict__ dWih0, const bf16* __restrict__ dWhh0,
    const bf16* __restrict__ dWih1, const bf16* __restrict__ dWhh1,
    const float* __restrict__ eb0, const float* __restrict__ eb1,
    const float* __restrict__ db0, const float* __restrict__ db1,
    const float* __restrict__ c0_in,
    bf16* __restrict__ h0buf, bf16* __restrict__ h1buf, // [2][B_][H_] each
    const bf16* __restrict__ Wlin, const float* __restrict__ blin,
    float* __restrict__ out, unsigned* __restrict__ barCnt) {
  const int lane  = threadIdx.x & 31;
  const int wgid  = blockIdx.x * (NTHREADS / 32) + (threadIdx.x >> 5);
  const int layer = wgid >> 8;   // 0 or 1
  const int task  = wgid & 255;
  const int mtile = task >> 6;   // 0..3   (batch tile)
  const int htile = task & 63;   // 0..63  (hidden tile)
  const int nn = lane & 15, hi = lane >> 4;
  unsigned gen = 0;

  // Cell state patch in registers (WMMA C/D layout), for entire enc+dec run.
  f32x8 c;
  {
    const float* cp = c0_in + (size_t)layer * B_ * H_ +
                      (size_t)(mtile * 16 + hi * 8) * H_ + htile * 16 + nn;
#pragma unroll
    for (int v = 0; v < 8; ++v) c[v] = cp[(size_t)v * H_];
  }

  for (int phase = 0; phase < 2; ++phase) {  // 0 = encoder, 1 = decoder
    const bf16* Wih0 = phase ? dWih0 : eWih0;
    const bf16* Whh0 = phase ? dWhh0 : eWhh0;
    const bf16* Wih1 = phase ? dWih1 : eWih1;
    const bf16* Whh1 = phase ? dWhh1 : eWhh1;
    const float* bias = layer ? (phase ? db1 : eb1) : (phase ? db0 : eb0);

    for (int s = 0; s <= T_; ++s) {
      const int pr = (s + 1) & 1;  // read parity  ((s-1)&1)
      const int pw = s & 1;        // write parity
      const bool active = (layer == 0) ? (s < T_) : (s >= 1);
      if (active) {
        f32x8 acc[4] = {};
        if (layer == 0) {
          // x_t = input_seq[:, t=s, :]  (bf16, row stride T*I, K=I)
          gemm4(acc, Xbf + (size_t)s * I_, (size_t)T_ * I_, Wih0, I_, I_,
                mtile, htile, lane);
          gemm4(acc, h0buf + (size_t)pr * B_ * H_, H_, Whh0, H_, H_,
                mtile, htile, lane);
        } else {
          // x = layer0 h at t=s-1; h_prev = layer1 h at t=s-2
          gemm4(acc, h0buf + (size_t)pr * B_ * H_, H_, Wih1, H_, H_,
                mtile, htile, lane);
          gemm4(acc, h1buf + (size_t)pr * B_ * H_, H_, Whh1, H_, H_,
                mtile, htile, lane);
        }
        const int bcol = htile * 16 + nn;
        const float bi = bias[0 * H_ + bcol];
        const float bf = bias[1 * H_ + bcol];
        const float bg = bias[2 * H_ + bcol];
        const float bo = bias[3 * H_ + bcol];
        bf16* hb = (layer ? h1buf : h0buf) + (size_t)pw * B_ * H_;
#pragma unroll
        for (int v = 0; v < 8; ++v) {
          const float ig = sigm(acc[0][v] + bi);
          const float fg = sigm(acc[1][v] + bf);
          const float gg = tanhf(acc[2][v] + bg);
          const float og = sigm(acc[3][v] + bo);
          const float cv = fg * c[v] + ig * gg;
          c[v] = cv;
          const float hv = og * tanhf(cv);
          hb[(size_t)(mtile * 16 + hi * 8 + v) * H_ + htile * 16 + nn] =
              (bf16)hv;
        }
      }
      grid_barrier(barCnt, gen);
    }
  }

  // Final linear: out = h1_final @ Wlin^T + blin. Decoder top-layer h at
  // t=T-1 was written at super-step s=T (even) -> parity 0 buffer.
  if (layer == 1 && htile < (O_ / 16)) {
    f32x8 acc = {};
    for (int k = 0; k < H_; k += 32) {
      bf16x16 a = load_a_frag(h1buf, H_, mtile * 16, k, lane);
      bf16x16 b = load_b_frag(Wlin, H_, htile * 16, k, lane);
      acc = wmma_bf16(a, b, acc);
    }
    const float bl = blin[htile * 16 + nn];
#pragma unroll
    for (int v = 0; v < 8; ++v)
      out[(size_t)(mtile * 16 + hi * 8 + v) * O_ + htile * 16 + nn] =
          acc[v] + bl;
  }
}

// ---------------------------------------------------------------------------
// Prologue kernels: f32 -> bf16 conversions + state/barrier init (per launch,
// deterministic, graph-capture safe).
// ---------------------------------------------------------------------------
__global__ void cvt_f32_bf16(const float* __restrict__ src,
                             bf16* __restrict__ dst, int n) {
  int i = blockIdx.x * blockDim.x + threadIdx.x;
  const int stride = gridDim.x * blockDim.x;
  for (; i < n; i += stride) dst[i] = (bf16)src[i];
}

__global__ void init_state(const float* __restrict__ h0in,
                           bf16* __restrict__ h0buf, bf16* __restrict__ h1buf,
                           unsigned* __restrict__ cnt) {
  int i = blockIdx.x * blockDim.x + threadIdx.x;
  if (i == 0) *cnt = 0;  // reset grid barrier each launch
  const int stride = gridDim.x * blockDim.x;
  for (; i < B_ * H_; i += stride) {
    h0buf[B_ * H_ + i] = (bf16)h0in[i];            // layer0 init -> parity 1
    h1buf[i]           = (bf16)h0in[B_ * H_ + i];  // layer1 init -> parity 0
  }
}

extern "C" void kernel_launch(void* const* d_in, const int* in_sizes, int n_in,
                              void* d_out, int out_size, void* d_ws,
                              size_t ws_size, hipStream_t stream) {
  (void)in_sizes; (void)n_in; (void)out_size; (void)ws_size;
  const float* input_seq = (const float*)d_in[0];
  const float* h0    = (const float*)d_in[1];
  const float* c0    = (const float*)d_in[2];
  const float* eWih0 = (const float*)d_in[3];
  const float* eWhh0 = (const float*)d_in[4];
  const float* eb0   = (const float*)d_in[5];
  const float* eWih1 = (const float*)d_in[6];
  const float* eWhh1 = (const float*)d_in[7];
  const float* eb1   = (const float*)d_in[8];
  const float* dWih0 = (const float*)d_in[9];
  const float* dWhh0 = (const float*)d_in[10];
  const float* db0   = (const float*)d_in[11];
  const float* dWih1 = (const float*)d_in[12];
  const float* dWhh1 = (const float*)d_in[13];
  const float* db1   = (const float*)d_in[14];
  const float* Wlin  = (const float*)d_in[15];
  const float* blin  = (const float*)d_in[16];
  float* out = (float*)d_out;

  char* ws = (char*)d_ws;
  size_t off = 0;
  auto take = [&](size_t bytes) -> void* {
    void* p = ws + off;
    off = (off + bytes + 255) & ~(size_t)255;
    return p;
  };
  unsigned* cnt = (unsigned*)take(256);
  bf16* Xbf    = (bf16*)take((size_t)B_ * T_ * I_ * 2);
  bf16* eWih0b = (bf16*)take((size_t)4 * H_ * I_ * 2);
  bf16* eWhh0b = (bf16*)take((size_t)4 * H_ * H_ * 2);
  bf16* eWih1b = (bf16*)take((size_t)4 * H_ * H_ * 2);
  bf16* eWhh1b = (bf16*)take((size_t)4 * H_ * H_ * 2);
  bf16* dWih0b = (bf16*)take((size_t)4 * H_ * I_ * 2);
  bf16* dWhh0b = (bf16*)take((size_t)4 * H_ * H_ * 2);
  bf16* dWih1b = (bf16*)take((size_t)4 * H_ * H_ * 2);
  bf16* dWhh1b = (bf16*)take((size_t)4 * H_ * H_ * 2);
  bf16* Wlinb  = (bf16*)take((size_t)O_ * H_ * 2);
  bf16* h0buf  = (bf16*)take((size_t)2 * B_ * H_ * 2);
  bf16* h1buf  = (bf16*)take((size_t)2 * B_ * H_ * 2);

  const int cb = 256, cg = 1024;
  cvt_f32_bf16<<<cg, cb, 0, stream>>>(input_seq, Xbf, B_ * T_ * I_);
  cvt_f32_bf16<<<cg, cb, 0, stream>>>(eWih0, eWih0b, 4 * H_ * I_);
  cvt_f32_bf16<<<cg, cb, 0, stream>>>(eWhh0, eWhh0b, 4 * H_ * H_);
  cvt_f32_bf16<<<cg, cb, 0, stream>>>(eWih1, eWih1b, 4 * H_ * H_);
  cvt_f32_bf16<<<cg, cb, 0, stream>>>(eWhh1, eWhh1b, 4 * H_ * H_);
  cvt_f32_bf16<<<cg, cb, 0, stream>>>(dWih0, dWih0b, 4 * H_ * I_);
  cvt_f32_bf16<<<cg, cb, 0, stream>>>(dWhh0, dWhh0b, 4 * H_ * H_);
  cvt_f32_bf16<<<cg, cb, 0, stream>>>(dWih1, dWih1b, 4 * H_ * H_);
  cvt_f32_bf16<<<cg, cb, 0, stream>>>(dWhh1, dWhh1b, 4 * H_ * H_);
  cvt_f32_bf16<<<cg, cb, 0, stream>>>(Wlin, Wlinb, O_ * H_);
  init_state<<<64, 256, 0, stream>>>(h0, h0buf, h1buf, cnt);

  lstm_seq2seq<<<NBLK, NTHREADS, 0, stream>>>(
      Xbf, eWih0b, eWhh0b, eWih1b, eWhh1b, dWih0b, dWhh0b, dWih1b, dWhh1b,
      eb0, eb1, db0, db1, c0, h0buf, h1buf, Wlinb, blin, out, cnt);
}
